// Expander_90615220011493
// MI455X (gfx1250) — matrix operand order repaired and checked by reading, not compiled, not run on hardware
//
#include <hip/hip_runtime.h>
#include <hip/hip_bf16.h>

// ---------------------------------------------------------------------------
// Shapes (fixed by the reference):
//   B=4, S=2048 -> T = 8192 tokens
//   rank = 256, d_model = 1024, n_expand = 16, n_reflect = 64, k = 2
//   GEMM: Y[T,1024] = Z[T,4096] @ E[4096,1024], Z[t, e*256+r] = w[t,e]*x[t,r]
//   E is pre-transposed+converted to Ebt[1024, 4096] bf16 so the GEMM's
//   B-fragment (two runs of 8 K per column) is two aligned 16B LDS reads.
// ---------------------------------------------------------------------------

typedef __attribute__((ext_vector_type(16))) __bf16 v16bf;
typedef __attribute__((ext_vector_type(8)))  __bf16 v8bf;
typedef __attribute__((ext_vector_type(8)))  float  v8f;

#define TOKENS   8192
#define RANK     256
#define DMODEL   1024
#define NEXP     16
#define NREFL    64
#define KDIM     4096   // NEXP * RANK

// ------------------ K0: transpose + convert E -> Ebt (bf16) -----------------
// E: (4096 k, 1024 d) f32  ->  Ebt: (1024 d, 4096 k) bf16
// 32x32 tiles through LDS: coalesced f32 reads, coalesced bf16 writes.
__global__ __launch_bounds__(256) void k_cvt_transpose(const float* __restrict__ E,
                                                       __bf16* __restrict__ Ebt) {
  __shared__ float tile[32][33];
  const int t = threadIdx.x;
  const int r = t >> 5;       // 0..7
  const int c = t & 31;       // 0..31
  const int k0 = blockIdx.y * 32;
  const int d0 = blockIdx.x * 32;

  #pragma unroll
  for (int i = 0; i < 4; ++i) {
    int kk = r + i * 8;
    tile[kk][c] = E[(size_t)(k0 + kk) * DMODEL + d0 + c];
  }
  __syncthreads();
  #pragma unroll
  for (int i = 0; i < 4; ++i) {
    int dd = r + i * 8;
    Ebt[(size_t)(d0 + dd) * KDIM + k0 + c] = (__bf16)tile[c][dd];
  }
}

// --------------- K1: router scores, top2, reflections, gate, Z --------------
__global__ __launch_bounds__(256) void k_router(const float* __restrict__ x,
                                                const float* __restrict__ Wr,
                                                const float* __restrict__ We,
                                                const float* __restrict__ reflect_r,
                                                int*   __restrict__ idxr_out,
                                                float* __restrict__ weights_out,
                                                __bf16* __restrict__ Z) {
  __shared__ float sx[RANK];
  __shared__ float sred[256];
  __shared__ float sred2[256];
  __shared__ float sscores[NREFL];
  __shared__ float sw[NEXP];
  __shared__ int   sidx[2];

  const int t   = threadIdx.x;
  const int tok = blockIdx.x;

  sx[t] = x[(size_t)tok * RANK + t];
  __syncthreads();

  // scores_r: 64 dots of length 256 (x from LDS broadcast, Wr L2-resident)
  if (t < NREFL) {
    const float* wrow = Wr + (size_t)t * RANK;
    float s = 0.f;
    #pragma unroll 8
    for (int r = 0; r < RANK; ++r) s += sx[r] * wrow[r];
    sscores[t] = s;
  }
  __syncthreads();

  // exact top-2 (JAX tie-break: lower index wins on ties -> strict '>')
  if (t == 0) {
    int b0 = 0; float v0 = sscores[0];
    for (int n = 1; n < NREFL; ++n)
      if (sscores[n] > v0) { v0 = sscores[n]; b0 = n; }
    int b1 = (b0 == 0) ? 1 : 0; float v1 = sscores[b1];
    for (int n = 0; n < NREFL; ++n)
      if (n != b0 && sscores[n] > v1) { v1 = sscores[n]; b1 = n; }
    sidx[0] = b0; sidx[1] = b1;
    idxr_out[(size_t)tok * 2 + 0] = b0;
    idxr_out[(size_t)tok * 2 + 1] = b1;
  }
  __syncthreads();

  // two reflections: x -= (2*<x,v>/(<v,v>+1e-8)) * v
  for (int kk = 0; kk < 2; ++kk) {
    const float* v = reflect_r + (size_t)sidx[kk] * RANK;
    const float vt = v[t];
    sred[t]  = sx[t] * vt;
    sred2[t] = vt * vt;
    __syncthreads();
    for (int s = 128; s > 0; s >>= 1) {
      if (t < s) { sred[t] += sred[t + s]; sred2[t] += sred2[t + s]; }
      __syncthreads();
    }
    const float c = 2.0f * sred[0] / (sred2[0] + 1e-8f);
    __syncthreads();
    sx[t] -= c * vt;
    __syncthreads();
  }

  // gate scores + softmax
  if (t < NEXP) {
    const float* wrow = We + (size_t)t * RANK;
    float s = 0.f;
    #pragma unroll 8
    for (int r = 0; r < RANK; ++r) s += sx[r] * wrow[r];
    sw[t] = s;
  }
  __syncthreads();
  if (t == 0) {
    float m = sw[0];
    for (int e = 1; e < NEXP; ++e) m = fmaxf(m, sw[e]);
    float sum = 0.f;
    for (int e = 0; e < NEXP; ++e) { float ex = expf(sw[e] - m); sw[e] = ex; sum += ex; }
    float inv = 1.0f / sum;
    for (int e = 0; e < NEXP; ++e) sw[e] *= inv;
  }
  __syncthreads();
  if (t < NEXP) weights_out[(size_t)tok * NEXP + t] = sw[t];

  // Z row: 4096 bf16, coalesced
  __bf16* zrow = Z + (size_t)tok * KDIM;
  #pragma unroll
  for (int i = 0; i < 16; ++i) {
    int idx = t + i * 256;
    int e = idx >> 8;
    int r = idx & 255;
    zrow[idx] = (__bf16)(sw[e] * sx[r]);
  }
}

// ----------------------- K2: WMMA GEMM  Y = Z @ Ebt^T -----------------------
#define GE_M 128
#define GE_N 64
#define GE_K 32
#define LDA  40   // A tile row stride in elements (128 x 32 bf16 + pad, 80B rows)
#define LDB  48   // B tile row stride in elements (64 n x 32 k bf16, 96B rows)

__global__ __launch_bounds__(256) void k_expand_gemm(const __bf16* __restrict__ Z,
                                                     const __bf16* __restrict__ Ebt,
                                                     float* __restrict__ Y) {
  __shared__ alignas(32) __bf16 sA[GE_M * LDA];
  __shared__ alignas(32) __bf16 sB[GE_N * LDB];

  const int t    = threadIdx.x;
  const int wave = t >> 5;
  const int lane = t & 31;
  const int m0 = blockIdx.y * GE_M;
  const int n0 = blockIdx.x * GE_N;

  v8f acc[4] = {};

  const int halfSel = (lane < 16) ? 0 : 1;
  const int aRow    = wave * 16 + (lane & 15);   // M row within tile
  const int aOffLo  = halfSel * 8;               // first 8 K values this lane owns
  const int bCol    = lane & 15;                 // N within 16-col subtile
  const int bOff    = halfSel * 8;               // first 8 K values this lane owns

  // A-tile addressing: 128x32 = 512 chunks of 8 bf16; this thread owns 2 chunks
  const int aRow0 = (t + 0)   >> 2, aKc0 = ((t + 0)   & 3) * 8;
  const int aRow1 = (t + 256) >> 2, aKc1 = ((t + 256) & 3) * 8;
  // B-tile addressing: 64x32 = 256 chunks of 8 bf16; this thread owns 1 chunk
  const int bRow  = t >> 2,        bKc  = (t & 3) * 8;

  for (int k0 = 0; k0 < KDIM; k0 += GE_K) {
    __syncthreads();
    {
      v8bf va0 = *(const v8bf*)(Z + (size_t)(m0 + aRow0) * KDIM + k0 + aKc0);
      v8bf va1 = *(const v8bf*)(Z + (size_t)(m0 + aRow1) * KDIM + k0 + aKc1);
      v8bf vb  = *(const v8bf*)(Ebt + (size_t)(n0 + bRow) * KDIM + k0 + bKc);
      *(v8bf*)(sA + aRow0 * LDA + aKc0) = va0;
      *(v8bf*)(sA + aRow1 * LDA + aKc1) = va1;
      *(v8bf*)(sB + bRow * LDB + bKc)   = vb;
    }
    // Prefetch next K-tile (one cacheline hint per row) to hide HBM latency
    // behind this tile's WMMAs; lowers to global_prefetch_b8.
    if (k0 + GE_K < KDIM) {
      if (t < GE_M)
        __builtin_prefetch(Z + (size_t)(m0 + t) * KDIM + k0 + GE_K, 0, 3);
      if (t < GE_N)
        __builtin_prefetch(Ebt + (size_t)(n0 + t) * KDIM + k0 + GE_K, 0, 3);
    }
    __syncthreads();

    // A fragment per gfx1250 wave32 layout:
    //  lanes 0-15 : row m=lane,    elems = K[0..7],  K[16..23]
    //  lanes 16-31: row m=lane-16, elems = K[8..15], K[24..31]
    v8bf alo = *(const v8bf*)(sA + aRow * LDA + aOffLo);
    v8bf ahi = *(const v8bf*)(sA + aRow * LDA + aOffLo + 16);
    v16bf a = __builtin_shufflevector(alo, ahi,
        0, 1, 2, 3, 4, 5, 6, 7, 8, 9, 10, 11, 12, 13, 14, 15);

    // 4 N-subtiles: B fragment = K[bOff..bOff+7], K[bOff+16..bOff+23] for column bCol
    #pragma unroll
    for (int s = 0; s < 4; ++s) {
      v8bf blo = *(const v8bf*)(sB + (s * 16 + bCol) * LDB + bOff);
      v8bf bhi = *(const v8bf*)(sB + (s * 16 + bCol) * LDB + bOff + 16);
      v16bf b = __builtin_shufflevector(blo, bhi,
          0, 1, 2, 3, 4, 5, 6, 7, 8, 9, 10, 11, 12, 13, 14, 15);
      acc[s] = __builtin_amdgcn_wmma_f32_16x16x32_bf16(
          false, a, false, b, (short)0, acc[s], false, false);
    }
  }

  // C/D layout: VGPR i -> (lanes 0-15: M=i, N=lane) (lanes 16-31: M=8+i, N=lane-16)
  #pragma unroll
  for (int s = 0; s < 4; ++s) {
    #pragma unroll
    for (int i = 0; i < 8; ++i) {
      int row = m0 + wave * 16 + (halfSel ? (8 + i) : i);
      int col = n0 + s * 16 + bCol;
      Y[(size_t)row * DMODEL + col] = acc[s][i];
    }
  }
}

// --------------- K3: decoder scores, top2, reflections, output --------------
__global__ __launch_bounds__(256) void k_decode(const float* __restrict__ Y,
                                                const float* __restrict__ Wd,
                                                const float* __restrict__ reflect_d,
                                                int*   __restrict__ idxd_out,
                                                float* __restrict__ xout) {
  __shared__ float sy[DMODEL];
  __shared__ float sred[256];
  __shared__ float sred2[256];
  __shared__ float sscores[NREFL];
  __shared__ int   sidx[2];

  const int t   = threadIdx.x;
  const int tok = blockIdx.x;

  #pragma unroll
  for (int i = 0; i < 4; ++i)
    sy[t + i * 256] = Y[(size_t)tok * DMODEL + t + i * 256];
  __syncthreads();

  if (t < NREFL) {
    const float* wrow = Wd + (size_t)t * DMODEL;
    float s = 0.f;
    #pragma unroll 8
    for (int d = 0; d < DMODEL; ++d) s += sy[d] * wrow[d];
    sscores[t] = s;
  }
  __syncthreads();

  if (t == 0) {
    int b0 = 0; float v0 = sscores[0];
    for (int n = 1; n < NREFL; ++n)
      if (sscores[n] > v0) { v0 = sscores[n]; b0 = n; }
    int b1 = (b0 == 0) ? 1 : 0; float v1 = sscores[b1];
    for (int n = 0; n < NREFL; ++n)
      if (n != b0 && sscores[n] > v1) { v1 = sscores[n]; b1 = n; }
    sidx[0] = b0; sidx[1] = b1;
    idxd_out[(size_t)tok * 2 + 0] = b0;
    idxd_out[(size_t)tok * 2 + 1] = b1;
  }
  __syncthreads();

  for (int kk = 0; kk < 2; ++kk) {
    const float* v = reflect_d + (size_t)sidx[kk] * DMODEL;
    float p = 0.f, q = 0.f;
    float vloc[4];
    #pragma unroll
    for (int i = 0; i < 4; ++i) {
      vloc[i] = v[t + i * 256];
      p += sy[t + i * 256] * vloc[i];
      q += vloc[i] * vloc[i];
    }
    sred[t] = p; sred2[t] = q;
    __syncthreads();
    for (int s = 128; s > 0; s >>= 1) {
      if (t < s) { sred[t] += sred[t + s]; sred2[t] += sred2[t + s]; }
      __syncthreads();
    }
    const float c = 2.0f * sred[0] / (sred2[0] + 1e-8f);
    __syncthreads();
    #pragma unroll
    for (int i = 0; i < 4; ++i) sy[t + i * 256] -= c * vloc[i];
    __syncthreads();
  }

  #pragma unroll
  for (int i = 0; i < 4; ++i)
    xout[(size_t)tok * DMODEL + t + i * 256] = sy[t + i * 256];
}

// ---------------------------------------------------------------------------
extern "C" void kernel_launch(void* const* d_in, const int* in_sizes, int n_in,
                              void* d_out, int out_size, void* d_ws, size_t ws_size,
                              hipStream_t stream) {
  (void)in_sizes; (void)n_in; (void)out_size; (void)ws_size;

  const float* x         = (const float*)d_in[0];  // (4,2048,256)
  const float* Wr        = (const float*)d_in[1];  // (64,256)
  const float* We        = (const float*)d_in[2];  // (16,256)
  const float* Wd        = (const float*)d_in[3];  // (64,1024)
  const float* E         = (const float*)d_in[4];  // (16,256,1024)
  const float* reflect_r = (const float*)d_in[5];  // (64,256)
  const float* reflect_d = (const float*)d_in[6];  // (64,1024)
  // d_in[7] = reflect_k (==2), compile-time constant here

  // d_out: x_out(8388608 f32) | indices_r(16384 i32) | weights_e(131072 f32) | indices_d(16384 i32)
  float* out      = (float*)d_out;
  float* xout     = out;
  int*   idxr_out = (int*)(out + 8388608);
  float* we_out   = out + 8388608 + 16384;
  int*   idxd_out = (int*)(out + 8388608 + 16384 + 131072);

  // workspace: Ebt (bf16, 1024x4096) | Z (bf16, 8192x4096) | Y (f32, 8192x1024)
  char* ws = (char*)d_ws;
  __bf16* Ebt = (__bf16*)ws;                                        //  8 MB
  __bf16* Z   = (__bf16*)(ws + (size_t)DMODEL * KDIM * 2);          // 64 MB
  float*  Y   = (float*)(ws + (size_t)DMODEL * KDIM * 2
                            + (size_t)TOKENS * KDIM * 2);           // 32 MB

  dim3 tgrid(DMODEL / 32, KDIM / 32);        // (32, 128)
  k_cvt_transpose<<<tgrid, 256, 0, stream>>>(E, Ebt);
  k_router<<<TOKENS, 256, 0, stream>>>(x, Wr, We, reflect_r, idxr_out, we_out, Z);
  dim3 ggrid(DMODEL / GE_N, TOKENS / GE_M);  // (16, 64)
  k_expand_gemm<<<ggrid, 256, 0, stream>>>(Z, Ebt, Y);
  k_decode<<<TOKENS, 256, 0, stream>>>(Y, Wd, reflect_d, idxd_out, xout);
}
